// ConvNAT_7189775254123
// MI455X (gfx1250) — matrix operand order repaired and checked by reading, not compiled, not run on hardware
//
#include <hip/hip_runtime.h>

// ---------------------------------------------------------------------------
// ConvNAT pipeline for MI455X (gfx1250, wave32, WMMA bf16 16x16x32).
// Stages: GroupNorm -> 3x3 conv (implicit GEMM) + GELU -> QKV 1x1 GEMMs ->
//         7x7 neighborhood attention (QK^T WMMA, reg softmax, PV WMMA).
// All GEMMs use v_wmma_f32_16x16x32_bf16 with fp32 accumulation.
// Round 2: batch all fragment loads ahead of each WMMA group so loads are
// clause'd and overlap WMMA execution instead of s_wait_loadcnt 0 per op.
// ---------------------------------------------------------------------------

typedef __attribute__((ext_vector_type(16))) __bf16   v16bf;
typedef __attribute__((ext_vector_type(8)))  float    v8f;
typedef __attribute__((ext_vector_type(4)))  unsigned int u32x4;

union Frag16 { v16bf v; u32x4 q[2]; unsigned short s[16]; };

#define DEVI __device__ __forceinline__

DEVI unsigned short f2bf(float f) {
    unsigned u = __float_as_uint(f);
    unsigned r = u + 0x7fffu + ((u >> 16) & 1u);   // round-to-nearest-even
    return (unsigned short)(r >> 16);
}

// geometry
#define NB   2
#define CIN  64
#define CMID 128
#define HH   96
#define WWD  96
#define HP   98           // padded H for conv input
#define WP   98
#define TPB  576          // tiles per batch image: 96 rows * 6 x-tiles
#define NTILES 1152       // NB * TPB
#define NATK 224          // attention key-patch size: 7 rows * 32 cols

// workspace layout (bytes)
#define OFF_XNP  ((size_t)0)                         // bf16 [2][98][98][64]
#define XNP_BYTES ((size_t)NB*HP*WP*CIN*2)           // 2,458,624
#define OFF_WB   (OFF_XNP + XNP_BYTES)               // bf16 [128][3][3][64]
#define WB_BYTES ((size_t)CMID*9*CIN*2)
#define OFF_WQ   (OFF_WB + WB_BYTES)
#define WQ_BYTES ((size_t)CMID*CMID*2)
#define OFF_WK   (OFF_WQ + WQ_BYTES)
#define OFF_WV   (OFF_WK + WQ_BYTES)
#define OFF_F    (OFF_WV + WQ_BYTES)                 // bf16 NHWC [2][96][96][128]
#define F_BYTES  ((size_t)NB*HH*WWD*CMID*2)
#define OFF_Q    (OFF_F + F_BYTES)
#define OFF_K    (OFF_Q + F_BYTES)
#define OFF_V    (OFF_K + F_BYTES)                   // bf16 CHW [2][128][96][96] (+pad)

DEVI void decode_tile(int tile, int& b, int& y, int& x0) {
    b = tile / TPB;
    int t = tile % TPB;
    y = t / 6;
    x0 = (t % 6) * 16;
}

// ---------------------------------------------------------------------------
// Weight prep: conv_w [co][ci][3][3] f32 -> wB [co][ky][kx][ci] bf16,
//              q/k/v_w [o][c] f32 -> bf16.
// ---------------------------------------------------------------------------
__global__ void k_prep(const float* __restrict__ conv_w,
                       const float* __restrict__ qw, const float* __restrict__ kw,
                       const float* __restrict__ vw,
                       unsigned short* __restrict__ wB,
                       unsigned short* __restrict__ wq, unsigned short* __restrict__ wk,
                       unsigned short* __restrict__ wv) {
    int idx = blockIdx.x * 256 + threadIdx.x;
    if (idx < CMID * 9 * CIN) {
        int co = idx / 576, k = idx % 576;
        int ky = k / 192, kx = (k % 192) / 64, ci = k & 63;
        wB[idx] = f2bf(conv_w[((co * CIN + ci) * 3 + ky) * 3 + kx]);
    } else {
        int i2 = idx - CMID * 9 * CIN;
        if (i2 < 3 * CMID * CMID) {
            int sel = i2 >> 14, o = i2 & 16383;
            const float* s = sel == 0 ? qw : (sel == 1 ? kw : vw);
            unsigned short* d = sel == 0 ? wq : (sel == 1 ? wk : wv);
            d[o] = f2bf(s[o]);
        }
    }
}

// ---------------------------------------------------------------------------
// GroupNorm: 16 groups of 4 channels; one block per (b,g). Writes padded
// NHWC bf16 (border pre-zeroed by memset).
// ---------------------------------------------------------------------------
__global__ void k_gn(const float* __restrict__ x, const float* __restrict__ gw,
                     const float* __restrict__ gb, unsigned short* __restrict__ xnp) {
    __shared__ float r0[256], r1[256];
    int b = blockIdx.x >> 4, g = blockIdx.x & 15;
    const float* xb = x + (size_t)(b * CIN + g * 4) * (HH * WWD);
    float s = 0.f, ss = 0.f;
    for (int i = threadIdx.x; i < 4 * HH * WWD; i += 256) {
        float v = xb[i]; s += v; ss += v * v;
    }
    r0[threadIdx.x] = s; r1[threadIdx.x] = ss; __syncthreads();
    for (int st = 128; st > 0; st >>= 1) {
        if ((int)threadIdx.x < st) { r0[threadIdx.x] += r0[threadIdx.x + st];
                                     r1[threadIdx.x] += r1[threadIdx.x + st]; }
        __syncthreads();
    }
    float mean = r0[0] * (1.f / 36864.f);
    float var  = r1[0] * (1.f / 36864.f) - mean * mean;
    float rstd = rsqrtf(var + 1e-5f);
    for (int i = threadIdx.x; i < 4 * HH * WWD; i += 256) {
        int cc = i / (HH * WWD), rem = i % (HH * WWD);
        int y = rem / WWD, xx = rem % WWD;
        int c = g * 4 + cc;
        float v = (xb[i] - mean) * rstd * gw[c] + gb[c];
        xnp[(((size_t)b * HP + (y + 1)) * WP + (xx + 1)) * CIN + c] = f2bf(v);
    }
}

// ---------------------------------------------------------------------------
// 3x3 conv as implicit GEMM: M = 16 pixels (row y, x0..x0+15), N = 128,
// K = 576 = 18 chunks of 32 (9 taps x 2 ci-halves). Epilogue: bias + exact GELU.
// One wave per M-tile, 4 waves per block. Loads batched ahead of WMMA group.
// ---------------------------------------------------------------------------
__global__ void __launch_bounds__(128) k_conv(const unsigned short* __restrict__ xnp,
                                              const unsigned short* __restrict__ wB,
                                              const float* __restrict__ cb,
                                              unsigned short* __restrict__ f) {
    int lane = threadIdx.x & 31, wid = threadIdx.x >> 5;
    int tile = blockIdx.x * 4 + wid;
    int b, y, x0; decode_tile(tile, b, y, x0);
    int lm = lane & 15, h = lane >> 4;

    v8f acc[8];
    v8f z = {0.f, 0.f, 0.f, 0.f, 0.f, 0.f, 0.f, 0.f};
#pragma unroll
    for (int nt = 0; nt < 8; nt++) acc[nt] = z;

    for (int kk = 0; kk < 18; kk++) {
        int ky = kk / 6, kx = (kk >> 1) % 3, ch = kk & 1;
        // batch all loads for this K-chunk, then run all 8 WMMAs
        Frag16 a;
        const unsigned short* ap =
            xnp + (((size_t)b * HP + (y + ky)) * WP + (x0 + lm + kx)) * CIN + ch * 32 + h * 8;
        a.q[0] = *(const u32x4*)(ap);
        a.q[1] = *(const u32x4*)(ap + 16);
        Frag16 bb[8];
#pragma unroll
        for (int nt = 0; nt < 8; nt++) {
            const unsigned short* bp = wB + (size_t)(nt * 16 + lm) * 576 + kk * 32 + h * 8;
            bb[nt].q[0] = *(const u32x4*)(bp);
            bb[nt].q[1] = *(const u32x4*)(bp + 16);
        }
#pragma unroll
        for (int nt = 0; nt < 8; nt++)
            acc[nt] = __builtin_amdgcn_wmma_f32_16x16x32_bf16(
                false, a.v, false, bb[nt].v, (short)0, acc[nt], false, false);
    }
#pragma unroll
    for (int nt = 0; nt < 8; nt++) {
        int co = nt * 16 + lm;
        float bias = cb[co];
#pragma unroll
        for (int r = 0; r < 8; r++) {
            int m = r + 8 * h;
            float v = acc[nt][r] + bias;
            v = 0.5f * v * (1.0f + erff(v * 0.70710678118654752f));  // exact GELU
            f[((size_t)((b * HH + y) * WWD) + x0 + m) * CMID + co] = f2bf(v);
        }
    }
}

// ---------------------------------------------------------------------------
// QKV 1x1 projections: three 16x128 GEMMs (K=128). q,k -> NHWC bf16;
// v -> CHW bf16 (A-matrix layout for the PV GEMM). Loads batched per chunk.
// ---------------------------------------------------------------------------
__global__ void __launch_bounds__(128) k_qkv(const unsigned short* __restrict__ f,
                                             const unsigned short* __restrict__ wq,
                                             const unsigned short* __restrict__ wk,
                                             const unsigned short* __restrict__ wv,
                                             const float* __restrict__ qb,
                                             const float* __restrict__ kb,
                                             const float* __restrict__ vb,
                                             unsigned short* __restrict__ qn,
                                             unsigned short* __restrict__ kn,
                                             unsigned short* __restrict__ vchw) {
    int lane = threadIdx.x & 31, wid = threadIdx.x >> 5;
    int tile = blockIdx.x * 4 + wid;
    int b, y, x0; decode_tile(tile, b, y, x0);
    int lm = lane & 15, h = lane >> 4;
    v8f z = {0.f, 0.f, 0.f, 0.f, 0.f, 0.f, 0.f, 0.f};

#pragma unroll
    for (int proj = 0; proj < 3; proj++) {
        const unsigned short* w = proj == 0 ? wq : (proj == 1 ? wk : wv);
        const float* bias = proj == 0 ? qb : (proj == 1 ? kb : vb);
        v8f acc[8];
#pragma unroll
        for (int nt = 0; nt < 8; nt++) acc[nt] = z;
        for (int kk = 0; kk < 4; kk++) {
            Frag16 a;
            const unsigned short* ap =
                f + ((size_t)((b * HH + y) * WWD) + x0 + lm) * CMID + kk * 32 + h * 8;
            a.q[0] = *(const u32x4*)(ap);
            a.q[1] = *(const u32x4*)(ap + 16);
            Frag16 bb[8];
#pragma unroll
            for (int nt = 0; nt < 8; nt++) {
                const unsigned short* bp = w + (size_t)(nt * 16 + lm) * CMID + kk * 32 + h * 8;
                bb[nt].q[0] = *(const u32x4*)(bp);
                bb[nt].q[1] = *(const u32x4*)(bp + 16);
            }
#pragma unroll
            for (int nt = 0; nt < 8; nt++)
                acc[nt] = __builtin_amdgcn_wmma_f32_16x16x32_bf16(
                    false, a.v, false, bb[nt].v, (short)0, acc[nt], false, false);
        }
#pragma unroll
        for (int nt = 0; nt < 8; nt++) {
            int co = nt * 16 + lm;
            float bv = bias[co];
#pragma unroll
            for (int r = 0; r < 8; r++) {
                int m = r + 8 * h;
                unsigned short o = f2bf(acc[nt][r] + bv);
                size_t pix = (size_t)((b * HH + y) * WWD) + x0 + m;
                if (proj == 0)      qn[pix * CMID + co] = o;
                else if (proj == 1) kn[pix * CMID + co] = o;
                else vchw[(((size_t)b * CMID + co) * HH + y) * WWD + x0 + m] = o;
            }
        }
    }
}

// ---------------------------------------------------------------------------
// Neighborhood attention. One wave per 16-query tile (row y, x0..x0+15).
// Key patch: 7 rows (si..si+6) x 32 cols (x0-8..x0+23, 8-aligned) -> 224 keys.
// logits = Q x K^T (14 N-tiles, K=128), mask window, *= sqrt(128), softmax in
// registers (shfl_xor over 16-lane halves), P bf16 -> LDS, O^T = V^T x P^T.
// B fragments batched 7 at a time; PV P-fragments hoisted out of c-tile loop.
// ---------------------------------------------------------------------------
__global__ void __launch_bounds__(128) k_natt(const unsigned short* __restrict__ qn,
                                              const unsigned short* __restrict__ kn,
                                              const unsigned short* __restrict__ vchw,
                                              float* __restrict__ out) {
    __shared__ __align__(32) unsigned short P[4][16][NATK];  // 28 KB
    int lane = threadIdx.x & 31, wid = threadIdx.x >> 5;
    int tile = blockIdx.x * 4 + wid;
    int b, y, x0; decode_tile(tile, b, y, x0);
    int lm = lane & 15, h = lane >> 4;
    int si = y - 3; si = si < 0 ? 0 : (si > 89 ? 89 : si);

    v8f z = {0.f, 0.f, 0.f, 0.f, 0.f, 0.f, 0.f, 0.f};
    v8f acc[14];
#pragma unroll
    for (int nt = 0; nt < 14; nt++) acc[nt] = z;

    // ---- logits: D[m][n] = sum_c Q[m][c] * K[n][c]
    for (int kk = 0; kk < 4; kk++) {
        Frag16 a;
        const unsigned short* ap =
            qn + ((size_t)((b * HH + y) * WWD) + x0 + lm) * CMID + kk * 32 + h * 8;
        a.q[0] = *(const u32x4*)(ap);
        a.q[1] = *(const u32x4*)(ap + 16);
#pragma unroll
        for (int g = 0; g < 2; g++) {            // 14 N-tiles in 2 batches of 7
            Frag16 bb[7];
#pragma unroll
            for (int i = 0; i < 7; i++) {
                int nt = g * 7 + i;
                int n = nt * 16 + lm;
                int ar = n >> 5, j = n & 31;
                int yk = si + ar;                // always 0..95
                int xk = x0 - 8 + j;             // may be OOB -> clamp; masked later
                xk = xk < 0 ? 0 : (xk > 95 ? 95 : xk);
                const unsigned short* bp =
                    kn + ((size_t)((b * HH + yk) * WWD) + xk) * CMID + kk * 32 + h * 8;
                bb[i].q[0] = *(const u32x4*)(bp);
                bb[i].q[1] = *(const u32x4*)(bp + 16);
            }
#pragma unroll
            for (int i = 0; i < 7; i++)
                acc[g * 7 + i] = __builtin_amdgcn_wmma_f32_16x16x32_bf16(
                    false, a.v, false, bb[i].v, (short)0, acc[g * 7 + i], false, false);
        }
    }

    // ---- mask + scale (scale MULTIPLIES logits, NATTEN semantics)
    const float SC = 11.313708498984761f;        // sqrt(128)
#pragma unroll
    for (int nt = 0; nt < 14; nt++) {
        int n = nt * 16 + lm;
        int j = n & 31;
        int xj = x0 - 8 + j;                     // true key column (unclamped)
#pragma unroll
        for (int r = 0; r < 8; r++) {
            int m = r + 8 * h;
            int sjm = x0 + m - 3; sjm = sjm < 0 ? 0 : (sjm > 89 ? 89 : sjm);
            bool valid = (xj >= sjm) && (xj <= sjm + 6);
            acc[nt][r] = valid ? acc[nt][r] * SC : -1e30f;
        }
    }

    // ---- row softmax (row m lives in reg r=m%8 across the 16 lanes of half h)
#pragma unroll
    for (int r = 0; r < 8; r++) {
        float mx = -1e30f;
#pragma unroll
        for (int nt = 0; nt < 14; nt++) mx = fmaxf(mx, acc[nt][r]);
        mx = fmaxf(mx, __shfl_xor(mx, 1, 32));
        mx = fmaxf(mx, __shfl_xor(mx, 2, 32));
        mx = fmaxf(mx, __shfl_xor(mx, 4, 32));
        mx = fmaxf(mx, __shfl_xor(mx, 8, 32));
        float sm = 0.f;
#pragma unroll
        for (int nt = 0; nt < 14; nt++) {
            float e = __expf(acc[nt][r] - mx);
            acc[nt][r] = e; sm += e;
        }
        sm += __shfl_xor(sm, 1, 32);
        sm += __shfl_xor(sm, 2, 32);
        sm += __shfl_xor(sm, 4, 32);
        sm += __shfl_xor(sm, 8, 32);
        float inv = 1.f / sm;
        int m = r + 8 * h;
#pragma unroll
        for (int nt = 0; nt < 14; nt++)
            P[wid][m][nt * 16 + lm] = f2bf(acc[nt][r] * inv);
    }
    __syncthreads();

    // ---- O^T = V^T x P^T : M=c (8 tiles), N=query m, K=keys (7 chunks of 32)
    // P fragments depend only on kk -> load once, reuse across all 8 c-tiles.
    Frag16 pb[7];
#pragma unroll
    for (int kk = 0; kk < 7; kk++) {
        const unsigned short* pp = &P[wid][lm][kk * 32 + h * 8];
        pb[kk].q[0] = *(const u32x4*)(pp);
        pb[kk].q[1] = *(const u32x4*)(pp + 16);
    }
    int xk0 = x0 - 8 + h * 8;                    // j0 in {0,8}
    int xk1 = x0 + 8 + h * 8;                    // j0 in {16,24}
    xk0 = xk0 < 0 ? 0 : (xk0 > 88 ? 88 : xk0);   // clamps keep 8-alignment
    xk1 = xk1 < 0 ? 0 : (xk1 > 88 ? 88 : xk1);   // clamped runs have P==0
#pragma unroll 1
    for (int mt = 0; mt < 8; mt++) {
        int c = mt * 16 + lm;
        Frag16 va[7];
#pragma unroll
        for (int kk = 0; kk < 7; kk++) {
            const unsigned short* vrow =
                vchw + (((size_t)b * CMID + c) * HH + (si + kk)) * WWD;
            va[kk].q[0] = *(const u32x4*)(vrow + xk0);
            va[kk].q[1] = *(const u32x4*)(vrow + xk1);
        }
        v8f o = z;
#pragma unroll
        for (int kk = 0; kk < 7; kk++)
            o = __builtin_amdgcn_wmma_f32_16x16x32_bf16(
                false, va[kk].v, false, pb[kk].v, (short)0, o, false, false);
#pragma unroll
        for (int r = 0; r < 8; r++) {
            int cc = mt * 16 + r + 8 * h;
            out[(((size_t)b * CMID + cc) * HH + y) * WWD + x0 + lm] = o[r];
        }
    }
}

// ---------------------------------------------------------------------------
extern "C" void kernel_launch(void* const* d_in, const int* in_sizes, int n_in,
                              void* d_out, int out_size, void* d_ws, size_t ws_size,
                              hipStream_t stream) {
    const float* x      = (const float*)d_in[0];
    const float* gn_w   = (const float*)d_in[1];
    const float* gn_b   = (const float*)d_in[2];
    const float* conv_w = (const float*)d_in[3];
    const float* conv_b = (const float*)d_in[4];
    const float* q_w    = (const float*)d_in[5];
    const float* q_b    = (const float*)d_in[6];
    const float* k_w    = (const float*)d_in[7];
    const float* k_b    = (const float*)d_in[8];
    const float* v_w    = (const float*)d_in[9];
    const float* v_b    = (const float*)d_in[10];
    (void)in_sizes; (void)n_in; (void)out_size; (void)ws_size;

    char* ws = (char*)d_ws;
    unsigned short* xnp  = (unsigned short*)(ws + OFF_XNP);
    unsigned short* wB   = (unsigned short*)(ws + OFF_WB);
    unsigned short* wq   = (unsigned short*)(ws + OFF_WQ);
    unsigned short* wk   = (unsigned short*)(ws + OFF_WK);
    unsigned short* wv   = (unsigned short*)(ws + OFF_WV);
    unsigned short* f    = (unsigned short*)(ws + OFF_F);
    unsigned short* qn   = (unsigned short*)(ws + OFF_Q);
    unsigned short* kn   = (unsigned short*)(ws + OFF_K);
    unsigned short* vchw = (unsigned short*)(ws + OFF_V);

    hipMemsetAsync(xnp, 0, XNP_BYTES, stream);                       // zero conv halo
    k_prep<<<480, 256, 0, stream>>>(conv_w, q_w, k_w, v_w, wB, wq, wk, wv);
    k_gn<<<32, 256, 0, stream>>>(x, gn_w, gn_b, xnp);
    k_conv<<<NTILES / 4, 128, 0, stream>>>(xnp, wB, conv_b, f);
    k_qkv<<<NTILES / 4, 128, 0, stream>>>(f, wq, wk, wv, q_b, k_b, v_b, qn, kn, vchw);
    k_natt<<<NTILES / 4, 128, 0, stream>>>(qn, kn, vchw, (float*)d_out);
}
